// Network_55774445305871
// MI455X (gfx1250) — matrix operand order repaired and checked by reading, not compiled, not run on hardware
//
#include <hip/hip_runtime.h>

typedef __attribute__((ext_vector_type(16))) _Float16 v16h;
typedef __attribute__((ext_vector_type(8)))  float    v8f;

#define THETA 10.0f
// psp alpha kernel eps(t)=(t/10)e^{1-t/10}  ->  IIR: u=A1*y2, y2=r1*y2+y1, y1=r1*y1+z
#define R1  0.90483741803595952f   // exp(-1/10)
#define A1C 0.24596031111569496f   // exp(0.9)/10
// refractory nu(t)=-2*theta*(t/2)e^{1-t/2}  ->  rho=A2*w2, w2=r2*w2+w1, w1=r2*w1+s
#define R2  0.60653065971263342f   // exp(-1/2)
#define A2C -16.487212707001282f   // -10*exp(1/2)

// ---------------- fragment layouts (wave32; A/B use this code's contiguous k order, not the ISA table order) ------------
// A (16x32 f16): lane l (m=l&15, hi=l>>4) holds halves h: k = hi*16 + h
// B (32x16 f16): lane l (n=l&15, hi=l>>4) holds halves h: k = hi*16 + h
// C/D (16x16 f32): vgpr r, lane l: m = r + 8*(l>>4), n = l&15

__global__ void slayer_convert_w(const float* __restrict__ W, _Float16* __restrict__ frag,
                                 int O, int K, int KT) {
  int fi = blockIdx.x;                 // fragment index: mt*KT + kt
  int mt = fi / KT, kt = fi - mt * KT;
  _Float16* dst = frag + (size_t)fi * 512;
#pragma unroll
  for (int q = 0; q < 4; ++q) {
    int e    = q * 128 + threadIdx.x;  // 0..511 element within fragment
    int lane = e >> 4, h = e & 15;
    int hi   = lane >> 4;
    int c    = h >> 3, j = h & 7;
    int o    = mt * 16 + (lane & 15);
    int k    = kt * 32 + hi * 16 + h;
    float v  = (o < O && k < K) ? W[(size_t)o * K + k] : 0.0f;
    dst[e] = (_Float16)v;
  }
}

__global__ void slayer_convert_x(const float* __restrict__ X, _Float16* __restrict__ frag,
                                 int K, int T, int NT, int KT) {
  int fi = blockIdx.x;                 // ((b*NT + tt)*KT + kt)
  int kt = fi % KT;
  int tt = (fi / KT) % NT;
  int b  = fi / (KT * NT);
  _Float16* dst = frag + (size_t)fi * 512;
#pragma unroll
  for (int q = 0; q < 4; ++q) {
    int e    = q * 128 + threadIdx.x;
    int lane = e >> 4, h = e & 15;
    int hi   = lane >> 4;
    int t    = tt * 16 + (lane & 15);
    int k    = kt * 32 + hi * 16 + h;
    float v  = (t < T && k < K) ? X[((size_t)b * K + k) * T + t] : 0.0f;
    dst[e] = (_Float16)v;
  }
}

// ------------------------------ WMMA GEMM ------------------------------------
// grid = (NT/BN, MT/BM, B), block = 32 (one wave -> BM x BN tiles of 16x16).
// Per-stream pointers advance by a compile-time constant so the compiler folds
// unrolled K-steps into immediate load offsets and software-pipelines them.
template <int BM, int BN>
__global__ void slayer_gemm_wmma(const _Float16* __restrict__ Af,
                                 const _Float16* __restrict__ Bf,
                                 float* __restrict__ Z,
                                 int KT, int Mpad, int Tpad, int NT) {
  int nt0 = blockIdx.x * BN, mt0 = blockIdx.y * BM, b = blockIdx.z;
  int lane = threadIdx.x;
  const size_t tstride = (size_t)KT * 512;  // halves between adjacent tiles' frag streams

  const _Float16* ap[BM];
  const _Float16* bp[BN];
#pragma unroll
  for (int i = 0; i < BM; ++i) ap[i] = Af + ((size_t)mt0 + i) * tstride + lane * 16;
#pragma unroll
  for (int j = 0; j < BN; ++j) bp[j] = Bf + ((size_t)b * NT + nt0 + j) * tstride + lane * 16;

  v8f acc[BM][BN];
#pragma unroll
  for (int i = 0; i < BM; ++i)
#pragma unroll
    for (int j = 0; j < BN; ++j) acc[i][j] = (v8f){};

#pragma unroll 2
  for (int kt = 0; kt < KT; ++kt) {
    v16h a[BM], bm[BN];
#pragma unroll
    for (int i = 0; i < BM; ++i) { a[i] = *(const v16h*)ap[i]; ap[i] += 512; }
#pragma unroll
    for (int j = 0; j < BN; ++j) { bm[j] = *(const v16h*)bp[j]; bp[j] += 512; }
#pragma unroll
    for (int i = 0; i < BM; ++i)
#pragma unroll
      for (int j = 0; j < BN; ++j)
        acc[i][j] = __builtin_amdgcn_wmma_f32_16x16x32_f16(false, a[i], false, bm[j],
                                                           (short)0, acc[i][j], false, false);
  }

  int hi = lane >> 4, n = lane & 15;
#pragma unroll
  for (int i = 0; i < BM; ++i)
#pragma unroll
    for (int j = 0; j < BN; ++j) {
      int t0 = (nt0 + j) * 16 + n;
#pragma unroll
      for (int r = 0; r < 8; ++r) {
        int m = (mt0 + i) * 16 + r + 8 * hi;
        Z[((size_t)b * Mpad + m) * Tpad + t0] = acc[i][j][r];
      }
    }
}

// ------------------- fused PSP conv + LIF spike dynamics ---------------------
// both alpha kernels are exact 2-state IIR filters: 4 registers per neuron,
// no ring buffer, no truncation. Writes next layer's B-fragments (f16, zeros
// in padded K rows / padded T cols) and/or the final fp32 output.
__global__ void slayer_psp_spike(const float* __restrict__ Z, int O, int Mpad, int Tpad,
                                 _Float16* __restrict__ Sfrag, int KTnext, int NT,
                                 float* __restrict__ out, int Oout, int T) {
  int o = blockIdx.x * blockDim.x + threadIdx.x;
  int b = blockIdx.y;
  int limit = Sfrag ? (KTnext * 32) : Mpad;
  if (o >= limit) return;
  bool act = (o < O);

  // B-fragment addressing pieces for k = o
  int kt = o >> 5, kl = o & 31;
  int fhi = kl >> 4, fh = kl & 15;

  const float* zp = Z + ((size_t)b * Mpad + o) * Tpad;
  float y1 = 0.f, y2 = 0.f, w1 = 0.f, w2 = 0.f;
  for (int t = 0; t < T; ++t) {
    float z = act ? zp[t] : 0.0f;
    y2 = fmaf(R1, y2, y1);
    y1 = fmaf(R1, y1, z);
    float u = A1C * y2;
    w2 = fmaf(R2, w2, w1);
    w1 = R2 * w1;
    float v = fmaf(A2C, w2, u);
    float s = (act && v >= THETA) ? 1.0f : 0.0f;
    w1 += s;
    if (Sfrag) {
      int tt = t >> 4, n = t & 15;
      size_t off = ((((size_t)b * NT + tt) * KTnext + kt) << 9) + (size_t)(n + (fhi << 4)) * 16 + fh;
      Sfrag[off] = (_Float16)s;
    }
    if (out && o < Oout) out[((size_t)b * Oout + o) * T + t] = s;
  }
  if (Sfrag) {
    for (int t = T; t < Tpad; ++t) {   // zero the padded T columns
      int tt = t >> 4, n = t & 15;
      size_t off = ((((size_t)b * NT + tt) * KTnext + kt) << 9) + (size_t)(n + (fhi << 4)) * 16 + fh;
      Sfrag[off] = (_Float16)0.0f;
    }
  }
}

// ------------------------------- launch --------------------------------------
extern "C" void kernel_launch(void* const* d_in, const int* in_sizes, int n_in,
                              void* d_out, int out_size, void* d_ws, size_t ws_size,
                              hipStream_t stream) {
  const float* inp = (const float*)d_in[0];  // [8,16384,300]
  const float* W1  = (const float*)d_in[1];  // [410,16384]
  const float* W2  = (const float*)d_in[2];  // [240,410]
  const float* W3  = (const float*)d_in[3];  // [10,240]
  float* out = (float*)d_out;                // [8,10,300]

  // Shapes / padding (T padded to 320 so NT=20 divides by BN=2)
  const int B = 8, T = 300, Tpad = 320, NT = 20;
  const int K1 = 16384, KT1 = 512, O1 = 410, M1 = 416, MT1 = 26;  // 26 = 2*13
  const int K2 = 410,  KT2 = 13,  O2 = 240, M2 = 240, MT2 = 15;   // 15 = 3*5
  const int K3 = 240,  KT3 = 8,   O3 = 10,  M3 = 16,  MT3 = 1;

  char* ws = (char*)d_ws;
  size_t off = 0;
  auto take = [&](size_t bytes) { char* p = ws + off; off = (off + bytes + 1023) & ~(size_t)1023; return p; };
  _Float16* Wf1 = (_Float16*)take((size_t)MT1 * KT1 * 512 * 2);       // 13.6 MB
  _Float16* Wf2 = (_Float16*)take((size_t)MT2 * KT2 * 512 * 2);
  _Float16* Wf3 = (_Float16*)take((size_t)MT3 * KT3 * 512 * 2);
  _Float16* Xf1 = (_Float16*)take((size_t)B * NT * KT1 * 512 * 2);    // 83.9 MB
  float*    Z1  = (float*)take((size_t)B * M1 * Tpad * 4);
  _Float16* Sf1 = (_Float16*)take((size_t)B * NT * KT2 * 512 * 2);
  float*    Z2  = (float*)take((size_t)B * M2 * Tpad * 4);
  _Float16* Sf2 = (_Float16*)take((size_t)B * NT * KT3 * 512 * 2);
  float*    Z3  = (float*)take((size_t)B * M3 * Tpad * 4);

  // 1) one-time f16 fragment-layout conversion (working set stays L2-resident)
  slayer_convert_w<<<MT1 * KT1, 128, 0, stream>>>(W1, Wf1, O1, K1, KT1);
  slayer_convert_w<<<MT2 * KT2, 128, 0, stream>>>(W2, Wf2, O2, K2, KT2);
  slayer_convert_w<<<MT3 * KT3, 128, 0, stream>>>(W3, Wf3, O3, K3, KT3);
  slayer_convert_x<<<B * NT * KT1, 128, 0, stream>>>(inp, Xf1, K1, T, NT, KT1);

  // 2) layer 1: 2x2 register-blocked WMMA GEMM
  slayer_gemm_wmma<2, 2><<<dim3(NT / 2, MT1 / 2, B), 32, 0, stream>>>(Wf1, Xf1, Z1, KT1, M1, Tpad, NT);
  slayer_psp_spike<<<dim3((KT2 * 32 + 127) / 128, B), 128, 0, stream>>>(
      Z1, O1, M1, Tpad, Sf1, KT2, NT, nullptr, 0, T);

  // 3) layer 2: 3x2 blocking (15 M-tiles = 3*5)
  slayer_gemm_wmma<3, 2><<<dim3(NT / 2, MT2 / 3, B), 32, 0, stream>>>(Wf2, Sf1, Z2, KT2, M2, Tpad, NT);
  slayer_psp_spike<<<dim3((KT3 * 32 + 127) / 128, B), 128, 0, stream>>>(
      Z2, O2, M2, Tpad, Sf2, KT3, NT, nullptr, 0, T);

  // 4) layer 3 -> final spikes
  slayer_gemm_wmma<1, 2><<<dim3(NT / 2, MT3, B), 32, 0, stream>>>(Wf3, Sf2, Z3, KT3, M3, Tpad, NT);
  slayer_psp_spike<<<dim3(1, B), 32, 0, stream>>>(
      Z3, O3, M3, Tpad, nullptr, 0, NT, out, O3, T);
}
